// GCNClassifier_54400055771431
// MI455X (gfx1250) — matrix operand order repaired and checked by reading, not compiled, not run on hardware
//
#include <hip/hip_runtime.h>

#define N_NODES 50000
#define N_EDGES 800000
#define HID 128
#define BN_EPS 1e-5f

typedef __attribute__((ext_vector_type(16))) __bf16 v16bf;
typedef __attribute__((ext_vector_type(8)))  __bf16 v8bf;
typedef __attribute__((ext_vector_type(8)))  float  v8f;

// ---------------------------------------------------------------------------
// H = X @ W   (M x 128) @ (128 x 128), bf16 WMMA, f32 accumulate.
// Block = 256 threads = 8 waves; block covers 128 rows, wave covers 16 rows.
// ---------------------------------------------------------------------------
__global__ __launch_bounds__(256) void gemm_xw_kernel(const float* __restrict__ X,
                                                      const float* __restrict__ W,
                                                      float* __restrict__ H, int M) {
  __shared__ __align__(16) __bf16 sX[128 * 128];   // 32 KB: X tile, row-major [r][k]
  __shared__ __align__(16) __bf16 sWt[128 * 128];  // 32 KB: W transposed   [n][k]

  const int tid  = threadIdx.x;
  const int row0 = blockIdx.x * 128;

  // Stage W transposed (k-major per output column n -> contiguous K for B frags)
  for (int i = tid; i < 128 * 128; i += 256) {
    int k = i >> 7, n = i & 127;
    sWt[n * 128 + k] = (__bf16)W[i];
  }
  // Stage X tile (zero-fill out-of-range rows)
  for (int i = tid; i < 128 * 128; i += 256) {
    int r = i >> 7, c = i & 127;
    int gr = row0 + r;
    float v = (gr < M) ? X[(size_t)gr * 128 + c] : 0.0f;
    sX[i] = (__bf16)v;
  }
  __syncthreads();

  const int wave = tid >> 5;
  const int lane = tid & 31;
  const int g    = lane >> 4;       // lane group 0/1
  const int ln   = lane & 15;
  const int tileRow = row0 + wave * 16;     // global row base of this wave's tile
  const bool active = tileRow < M;          // wave-uniform (M % 16 == 0)
  const int lrow = wave * 16 + ln;          // local row in sX for A fragment

  // A fragments, 4 K-steps of 32.
  // ISA 16-bit A 16x32: lanes 0-15 row M=lane, elems 0..7 = K 0..7, 8..15 = K 16..23;
  // lanes 16-31 same rows, K offset +8.
  v16bf afrag[4];
#pragma unroll
  for (int kt = 0; kt < 4; ++kt) {
    int kb = kt * 32 + g * 8;
    v8bf lo = *(const v8bf*)&sX[lrow * 128 + kb];
    v8bf hi = *(const v8bf*)&sX[lrow * 128 + kb + 16];
#pragma unroll
    for (int e = 0; e < 8; ++e) { afrag[kt][e] = lo[e]; afrag[kt][e + 8] = hi[e]; }
  }

  // Sweep the 8 column tiles of the output.
#pragma unroll
  for (int nt = 0; nt < 8; ++nt) {
    const int n = nt * 16 + ln;          // output column for this lane
    v8f acc = {};
#pragma unroll
    for (int kt = 0; kt < 4; ++kt) {
      // ISA 16-bit B 32x16: lanes 0-15 elems 0..7 = K 0..7, 8..15 = K 16..23; lanes 16-31 K offset +8.
      int kb = kt * 32 + g * 8;
      v8bf lo = *(const v8bf*)&sWt[n * 128 + kb];
      v8bf hi = *(const v8bf*)&sWt[n * 128 + kb + 16];
      v16bf bfrag;
#pragma unroll
      for (int e = 0; e < 8; ++e) { bfrag[e] = lo[e]; bfrag[e + 8] = hi[e]; }
      acc = __builtin_amdgcn_wmma_f32_16x16x32_bf16(
          /*neg_a=*/false, afrag[kt], /*neg_b=*/false, bfrag,
          /*c_mod=*/(short)0, acc, /*reuse_a=*/false, /*reuse_b=*/false);
    }
    if (active) {
      // C/D layout: VGPR i -> M = i + 8*g, N = lane%16
#pragma unroll
      for (int i = 0; i < 8; ++i)
        H[(size_t)(tileRow + g * 8 + i) * 128 + n] = acc[i];
    }
  }
}

// ---------------------------------------------------------------------------
// Helpers
// ---------------------------------------------------------------------------
__global__ void zero_kernel(float* p, int n) {
  int t = blockIdx.x * blockDim.x + threadIdx.x;
  if (t < n) p[t] = 0.0f;
}

__global__ void degree_kernel(const long long* __restrict__ ei, float* __restrict__ deg) {
  int e = blockIdx.x * blockDim.x + threadIdx.x;
  if (e < N_EDGES) atomicAdd(&deg[(int)ei[(size_t)N_EDGES + e]], 1.0f);
}

__global__ void dinv_kernel(float* __restrict__ d) {
  int i = blockIdx.x * blockDim.x + threadIdx.x;
  if (i < N_NODES) d[i] = rsqrtf(d[i] + 1.0f);   // degree incl. self-loop
}

// agg := H * dinv^2 + bias   (self-loop term + bias, before edge scatter)
__global__ void init_agg_kernel(const float* __restrict__ H, const float* __restrict__ dinv,
                                const float* __restrict__ bias, float* __restrict__ agg) {
  int t = blockIdx.x * blockDim.x + threadIdx.x;
  if (t >= N_NODES * HID) return;
  int n = t >> 7, c = t & 127;
  float di = dinv[n];
  agg[t] = H[t] * di * di + bias[c];
}

// agg[dst] += H[src] * dinv[src]*dinv[dst]  (one thread per edge-channel)
__global__ __launch_bounds__(256) void scatter_kernel(const long long* __restrict__ ei,
                                                      const float* __restrict__ dinv,
                                                      const float* __restrict__ H,
                                                      float* __restrict__ agg) {
  long long t = (long long)blockIdx.x * blockDim.x + threadIdx.x;
  int e = (int)(t >> 7);
  int c = (int)(t & 127);
  if (e >= N_EDGES) return;
  int s = (int)ei[e];
  int d = (int)ei[(size_t)N_EDGES + e];
  float nrm = dinv[s] * dinv[d];
  atomicAdd(&agg[(size_t)d * 128 + c], H[(size_t)s * 128 + c] * nrm);
}

// Per-channel sum / sum-of-squares over the node dimension -> stats[0..127], stats[128..255]
__global__ __launch_bounds__(256) void stats_kernel(const float* __restrict__ V,
                                                    float* __restrict__ stats) {
  int tid = threadIdx.x;
  int c = tid & 127, half = tid >> 7;
  float s = 0.0f, sq = 0.0f;
  for (int n = blockIdx.x * 2 + half; n < N_NODES; n += gridDim.x * 2) {
    float v = V[(size_t)n * 128 + c];
    s += v; sq += v * v;
  }
  __shared__ float rs[256], rq[256];
  rs[tid] = s; rq[tid] = sq;
  __syncthreads();
  if (half == 0) {
    atomicAdd(&stats[c],       s  + rs[tid + 128]);
    atomicAdd(&stats[128 + c], sq + rq[tid + 128]);
  }
}

// In-place training-mode BatchNorm + ReLU
__global__ void bn_relu_kernel(float* __restrict__ V, const float* __restrict__ stats,
                               const float* __restrict__ gam, const float* __restrict__ bet) {
  int t = blockIdx.x * blockDim.x + threadIdx.x;
  if (t >= N_NODES * HID) return;
  int c = t & 127;
  const float inv_n = 1.0f / (float)N_NODES;
  float mean = stats[c] * inv_n;
  float var  = stats[128 + c] * inv_n - mean * mean;
  float x = (V[t] - mean) * rsqrtf(var + BN_EPS);
  x = gam[c] * x + bet[c];
  V[t] = fmaxf(x, 0.0f);
}

// Mean pool (from channel sums) + 2-layer MLP head; single block of 128 threads.
__global__ void classifier_kernel(const float* __restrict__ stats,
                                  const float* __restrict__ cW1, const float* __restrict__ cb1,
                                  const float* __restrict__ cW2, const float* __restrict__ cb2,
                                  float* __restrict__ out) {
  __shared__ float pooled[128];
  __shared__ float h[64];
  int t = threadIdx.x;
  if (t < 128) pooled[t] = stats[t] * (1.0f / (float)N_NODES);
  __syncthreads();
  if (t < 64) {
    float a = cb1[t];
    for (int c = 0; c < 128; ++c) a = fmaf(pooled[c], cW1[c * 64 + t], a);
    h[t] = fmaxf(a, 0.0f);
  }
  __syncthreads();
  if (t < 5) {
    float a = cb2[t];
    for (int j = 0; j < 64; ++j) a = fmaf(h[j], cW2[j * 5 + t], a);
    out[t] = a;
  }
}

// ---------------------------------------------------------------------------
extern "C" void kernel_launch(void* const* d_in, const int* in_sizes, int n_in,
                              void* d_out, int out_size, void* d_ws, size_t ws_size,
                              hipStream_t stream) {
  (void)in_sizes; (void)n_in; (void)out_size; (void)ws_size;
  const float*     x   = (const float*)d_in[0];
  const long long* ei  = (const long long*)d_in[1];
  const float* Wl[3] = {(const float*)d_in[2],  (const float*)d_in[6],  (const float*)d_in[10]};
  const float* bl[3] = {(const float*)d_in[3],  (const float*)d_in[7],  (const float*)d_in[11]};
  const float* gl[3] = {(const float*)d_in[4],  (const float*)d_in[8],  (const float*)d_in[12]};
  const float* el[3] = {(const float*)d_in[5],  (const float*)d_in[9],  (const float*)d_in[13]};
  const float* cW1 = (const float*)d_in[14];
  const float* cb1 = (const float*)d_in[15];
  const float* cW2 = (const float*)d_in[16];
  const float* cb2 = (const float*)d_in[17];
  float* out = (float*)d_out;

  const size_t NF = (size_t)N_NODES * HID;        // 6.4M floats
  float* P     = (float*)d_ws;                    // layer activations (agg/output)
  float* Q     = P + NF;                          // GEMM output H scratch
  float* dinv  = Q + NF;                          // [N_NODES]
  float* stats = dinv + N_NODES;                  // [256]

  const int gemmBlocks = (N_NODES + 127) / 128;                 // 391
  const int nfBlocks   = (int)((NF + 255) / 256);               // 25000
  const int scBlocks   = (int)(((size_t)N_EDGES * 128) / 256);  // 400000
  const int degBlocks  = (N_EDGES + 255) / 256;                 // 3125
  const int nBlocks    = (N_NODES + 255) / 256;

  // Degree normalization (edge-only; compute once)
  zero_kernel<<<nBlocks, 256, 0, stream>>>(dinv, N_NODES);
  degree_kernel<<<degBlocks, 256, 0, stream>>>(ei, dinv);
  dinv_kernel<<<nBlocks, 256, 0, stream>>>(dinv);

  const float* layer_in = x;
  for (int l = 0; l < 3; ++l) {
    gemm_xw_kernel<<<gemmBlocks, 256, 0, stream>>>(layer_in, Wl[l], Q, N_NODES);
    init_agg_kernel<<<nfBlocks, 256, 0, stream>>>(Q, dinv, bl[l], P);
    scatter_kernel<<<scBlocks, 256, 0, stream>>>(ei, dinv, Q, P);
    zero_kernel<<<1, 256, 0, stream>>>(stats, 256);
    stats_kernel<<<256, 256, 0, stream>>>(P, stats);
    bn_relu_kernel<<<nfBlocks, 256, 0, stream>>>(P, stats, gl[l], el[l]);
    layer_in = P;
  }

  // Mean pool + classifier head
  zero_kernel<<<1, 256, 0, stream>>>(stats, 256);
  stats_kernel<<<256, 256, 0, stream>>>(P, stats);
  classifier_kernel<<<1, 128, 0, stream>>>(stats, cW1, cb1, cW2, cb2, out);
}
